// HetroGAT_18537078849563
// MI455X (gfx1250) — compile-verified, matrix operand order checked by reading, not run on hardware
//
#include <hip/hip_runtime.h>
#include <hip/hip_bf16.h>
#include <math.h>

#define NNODES 50000
#define NEDGES 500000
#define HID    128
#define NHEAD  16
#define HDIM   8
#define BN_EPS 1e-5f

typedef __attribute__((ext_vector_type(16))) __bf16 v16bf;
typedef __attribute__((ext_vector_type(8)))  __bf16 v8bf;
typedef __attribute__((ext_vector_type(4)))  __bf16 v4bf;
typedef __attribute__((ext_vector_type(8)))  float  v8f;

// ---------------------------------------------------------------------------
// WMMA GEMM: C[M x NOUT] = op_a(A[M x K]) @ B[K x NOUT] (+bias) (+skip/accum)
//   AM: 0 = raw A, 1 = relu(A*scale+shift) (BN+ReLU), 2 = leaky-relu(A, 0.01)
//   CM: 0 = store, 1 = store + skip-add of transformed A[row,col] (NOUT==K),
//       2 = accumulate into existing C (residual)
// Block = NOUT/16 waves; each wave owns one 16x16 output tile; grid = M/16.
// B is staged once per block into LDS as bf16, transposed & padded (KP=K+8,
// row stride 2*KP bytes = multiple of 16 -> aligned, conflict-free b128 reads).
// A tile (16 x K) is staged once (transform fused) and shared by all waves.
// All trip counts exact -> no divergence, EXEC all-ones for V_WMMA.
// ---------------------------------------------------------------------------
template<int K, int NOUT, int AM, int CM>
__global__ __launch_bounds__(32 * (NOUT / 16))
void gemm_wmma(const float* __restrict__ A, const float* __restrict__ B,
               const float* __restrict__ bias,
               const float* __restrict__ scale, const float* __restrict__ shift,
               float* __restrict__ C)
{
    constexpr int KP = K + 8;          // padded bf16 row stride
    constexpr int NW = NOUT / 16;
    constexpr int NT = 32 * NW;
    __shared__ __bf16 Bt[NOUT * KP];   // B transposed: Bt[col*KP + k]
    __shared__ __bf16 As[16 * KP];     // transformed A tile: As[row*KP + k]

    const int tid = threadIdx.x;
    const int m0  = blockIdx.x * 16;

    // ---- stage B^T (bf16) into LDS: float4 global loads, scalar LDS stores
#pragma unroll
    for (int i = 0; i < (K * NOUT) / (NT * 4); ++i) {
        const int idx = (tid + i * NT) * 4;
        const int k = idx / NOUT, col = idx % NOUT;
        const float4 v = *(const float4*)(B + idx);
        Bt[(col + 0) * KP + k] = (__bf16)v.x;
        Bt[(col + 1) * KP + k] = (__bf16)v.y;
        Bt[(col + 2) * KP + k] = (__bf16)v.z;
        Bt[(col + 3) * KP + k] = (__bf16)v.w;
    }
    // ---- stage transformed A tile (bf16) into LDS: float4 loads, b64 stores
#pragma unroll
    for (int i = 0; i < (16 * K) / (NT * 4); ++i) {
        const int idx = (tid + i * NT) * 4;
        const int row = idx / K, k = idx % K;
        float4 v = *(const float4*)(A + (size_t)(m0 + row) * K + k);
        if (AM == 1) {
            float t;
            t = v.x * scale[k + 0] + shift[k + 0]; v.x = t > 0.f ? t : 0.f;
            t = v.y * scale[k + 1] + shift[k + 1]; v.y = t > 0.f ? t : 0.f;
            t = v.z * scale[k + 2] + shift[k + 2]; v.z = t > 0.f ? t : 0.f;
            t = v.w * scale[k + 3] + shift[k + 3]; v.w = t > 0.f ? t : 0.f;
        } else if (AM == 2) {
            v.x = v.x > 0.f ? v.x : 0.01f * v.x;
            v.y = v.y > 0.f ? v.y : 0.01f * v.y;
            v.z = v.z > 0.f ? v.z : 0.01f * v.z;
            v.w = v.w > 0.f ? v.w : 0.01f * v.w;
        }
        v4bf p; p[0] = (__bf16)v.x; p[1] = (__bf16)v.y;
                p[2] = (__bf16)v.z; p[3] = (__bf16)v.w;
        *(v4bf*)&As[row * KP + k] = p;
    }
    __syncthreads();

    const int wave = tid >> 5, lane = tid & 31;
    const int half = lane >> 4, l16 = lane & 15;
    const int colB = wave * 16 + l16;

    v8f c = {};
#pragma unroll
    for (int k0 = 0; k0 < K; k0 += 32) {
        // per-lane K map: half0 -> {0..7,16..23}, half1 -> {8..15,24..31}
        const int ka = k0 + half * 8;
        const v8bf alo = *(const v8bf*)&As[l16 * KP + ka];
        const v8bf ahi = *(const v8bf*)&As[l16 * KP + ka + 16];
        const v8bf blo = *(const v8bf*)&Bt[colB * KP + ka];
        const v8bf bhi = *(const v8bf*)&Bt[colB * KP + ka + 16];
        v16bf a, b;
#pragma unroll
        for (int e = 0; e < 8; ++e) {
            a[e] = alo[e]; a[8 + e] = ahi[e];
            b[e] = blo[e]; b[8 + e] = bhi[e];
        }
        c = __builtin_amdgcn_wmma_f32_16x16x32_bf16(
                /*neg_a=*/false, a, /*neg_b=*/false, b,
                /*c_mod=*/(short)0, c, /*reuse_a=*/false, /*reuse_b=*/false);
    }

    const float bv = bias ? bias[colB] : 0.f;
#pragma unroll
    for (int v = 0; v < 8; ++v) {
        const int row = m0 + v + half * 8;      // C/D layout: lanes>=16 hold M+8
        const size_t idx = (size_t)row * NOUT + colB;
        float r = c[v] + bv;
        if (CM == 1) r += (float)As[(v + half * 8) * KP + colB];  // skip-add
        if (CM == 2) r += C[idx];                                 // residual
        C[idx] = r;
    }
}

// ---------------- BatchNorm statistics (training mode, biased var) ----------
__global__ void zero_stats_kernel(float* __restrict__ stats)
{
    stats[threadIdx.x] = 0.f;   // 256 entries: sum[0..127], sumsq[128..255]
}

__global__ void colstats_kernel(const float* __restrict__ X, int Cdim,
                                float* __restrict__ stats, int n, int rpb)
{
    const int j  = threadIdx.x;            // column, blockDim.x == Cdim
    const int r0 = blockIdx.x * rpb;
    const int r1 = (r0 + rpb < n) ? (r0 + rpb) : n;
    float s = 0.f, sq = 0.f;
    for (int r = r0; r < r1; ++r) {
        float v = X[(size_t)r * Cdim + j];
        s += v; sq += v * v;
    }
    atomicAdd(&stats[j],       s);
    atomicAdd(&stats[128 + j], sq);
}

__global__ void bnfin_kernel(const float* __restrict__ stats,
                             const float* __restrict__ gamma,
                             const float* __restrict__ beta,
                             float invN,
                             float* __restrict__ scl, float* __restrict__ shf)
{
    const int j  = threadIdx.x;            // blockDim.x == Cdim
    float mu  = stats[j] * invN;
    float var = stats[128 + j] * invN - mu * mu;
    float rs  = rsqrtf(var + BN_EPS);
    float sc  = rs * gamma[j];
    scl[j] = sc;
    shf[j] = beta[j] - mu * sc;
}

// ---------------- GAT per-node attention logits -----------------------------
__global__ void elr_kernel(const float* __restrict__ z,
                           const float* __restrict__ al,
                           const float* __restrict__ ar,
                           float* __restrict__ el, float* __restrict__ er)
{
    const int t = blockIdx.x * blockDim.x + threadIdx.x;   // node*16 + head
    const int node = t >> 4, head = t & 15;
    const float* zr = z + (size_t)node * HID + head * HDIM;
    float sl = 0.f, sr = 0.f;
#pragma unroll
    for (int d = 0; d < HDIM; ++d) {
        float v = zr[d];
        sl += v * al[head * HDIM + d];
        sr += v * ar[head * HDIM + d];
    }
    el[t] = sl;
    er[t] = sr;
}

// ---------------- segment-max / exp-sum / aggregate -------------------------
__device__ __forceinline__ float atomicMaxFloat(float* addr, float val)
{
    if (val >= 0.f)
        return __int_as_float(atomicMax((int*)addr, __float_as_int(val)));
    else
        return __uint_as_float(atomicMin((unsigned int*)addr, __float_as_uint(val)));
}

__global__ void init_ms_kernel(float* __restrict__ m, float* __restrict__ s)
{
    const int t = blockIdx.x * blockDim.x + threadIdx.x;   // N*16 threads
    m[t] = -INFINITY;
    s[t] = 0.f;
}

__global__ void init_acc_kernel(float* __restrict__ acc,
                                const float* __restrict__ b0,
                                const float* __restrict__ b1,
                                const float* __restrict__ b2)
{
    const int t = blockIdx.x * blockDim.x + threadIdx.x;   // N*128 threads
    const int c = t & (HID - 1);
    acc[t] = b0[c] + b1[c] + b2[c];
}

__global__ void scores_max_kernel(const int* __restrict__ src,
                                  const int* __restrict__ dst,
                                  const float* __restrict__ el,
                                  const float* __restrict__ er,
                                  float* __restrict__ m)
{
    const int t = blockIdx.x * blockDim.x + threadIdx.x;   // E*16 threads
    const int e = t >> 4, head = t & 15;
    const int si = src[e], di = dst[e];
    float v = el[si * NHEAD + head] + er[di * NHEAD + head];
    v = v > 0.f ? v : 0.2f * v;
    atomicMaxFloat(&m[di * NHEAD + head], v);
}

__global__ void exp_sum_kernel(const int* __restrict__ src,
                               const int* __restrict__ dst,
                               const float* __restrict__ el,
                               const float* __restrict__ er,
                               const float* __restrict__ m,
                               float* __restrict__ s)
{
    const int t = blockIdx.x * blockDim.x + threadIdx.x;   // E*16 threads
    const int e = t >> 4, head = t & 15;
    const int si = src[e], di = dst[e];
    float v = el[si * NHEAD + head] + er[di * NHEAD + head];
    v = v > 0.f ? v : 0.2f * v;
    atomicAdd(&s[di * NHEAD + head], expf(v - m[di * NHEAD + head]));
}

// One wave per edge: lane covers 4 channels (128 total), head = lane/2.
__global__ void aggregate_kernel(const int* __restrict__ src,
                                 const int* __restrict__ dst,
                                 const float* __restrict__ el,
                                 const float* __restrict__ er,
                                 const float* __restrict__ m,
                                 const float* __restrict__ s,
                                 const float* __restrict__ z,
                                 float* __restrict__ acc)
{
    const int widx = threadIdx.x >> 5;
    const int e = blockIdx.x * 8 + widx;                   // 8 waves / block
    const int lane = threadIdx.x & 31;
    const int head = lane >> 1;
    const int c4 = lane * 4;
    const int si = src[e], di = dst[e];
    float v = el[si * NHEAD + head] + er[di * NHEAD + head];
    v = v > 0.f ? v : 0.2f * v;
    const float alpha = expf(v - m[di * NHEAD + head]) / s[di * NHEAD + head];
    const float4 zv = *(const float4*)(z + (size_t)si * HID + c4);
    float* ap = acc + (size_t)di * HID + c4;
    atomicAdd(ap + 0, alpha * zv.x);
    atomicAdd(ap + 1, alpha * zv.y);
    atomicAdd(ap + 2, alpha * zv.z);
    atomicAdd(ap + 3, alpha * zv.w);
}

// ---------------- decision head: bnrelu(tmp) @ w2[128x2] + b2 ---------------
__global__ void decision_kernel(const float* __restrict__ tmp,
                                const float* __restrict__ scl,
                                const float* __restrict__ shf,
                                const float* __restrict__ w2,
                                const float* __restrict__ b2,
                                float* __restrict__ out, int n)
{
    const int i = blockIdx.x * blockDim.x + threadIdx.x;
    if (i >= n) return;
    const float* row = tmp + (size_t)i * HID;
    float a0 = 0.f, a1 = 0.f;
#pragma unroll 4
    for (int k = 0; k < HID; ++k) {
        float t = row[k] * scl[k] + shf[k];
        t = t > 0.f ? t : 0.f;
        a0 += t * w2[k * 2 + 0];
        a1 += t * w2[k * 2 + 1];
    }
    out[i * 2 + 0] = a0 + b2[0];
    out[i * 2 + 1] = a1 + b2[1];
}

// ---------------------------------------------------------------------------
extern "C" void kernel_launch(void* const* d_in, const int* in_sizes, int n_in,
                              void* d_out, int out_size, void* d_ws, size_t ws_size,
                              hipStream_t stream)
{
    (void)in_sizes; (void)n_in; (void)out_size; (void)ws_size;
    // ---- inputs (setup_inputs order; params flattened in sorted-dict order)
    const float* x = (const float*)d_in[0];
    const int* src[3] = {(const int*)d_in[1], (const int*)d_in[3], (const int*)d_in[5]};
    const int* dst[3] = {(const int*)d_in[2], (const int*)d_in[4], (const int*)d_in[6]};
    // decision: b1,b2,beta,gamma,w1,w2
    const float* dec_b1 = (const float*)d_in[7];
    const float* dec_b2 = (const float*)d_in[8];
    const float* dec_be = (const float*)d_in[9];
    const float* dec_ga = (const float*)d_in[10];
    const float* dec_w1 = (const float*)d_in[11];
    const float* dec_w2 = (const float*)d_in[12];
    // embed: b1,b2,beta,gamma,w1,w2
    const float* emb_b1 = (const float*)d_in[13];
    const float* emb_b2 = (const float*)d_in[14];
    const float* emb_be = (const float*)d_in[15];
    const float* emb_ga = (const float*)d_in[16];
    const float* emb_w1 = (const float*)d_in[17];
    const float* emb_w2 = (const float*)d_in[18];
    // gat[l][r]: W,al,ar,b at 19 + (l*3+r)*4 ; mlp[l]: 55 + l*6

    // ---- workspace layout (floats)
    float* ws   = (float*)d_ws;
    float* h    = ws;                                   // N*128
    float* tmp  = h    + (size_t)NNODES * HID;          // N*128
    float* z    = tmp  + (size_t)NNODES * HID;          // N*128
    float* acc  = z    + (size_t)NNODES * HID;          // N*128
    float* el   = acc  + (size_t)NNODES * HID;          // N*16
    float* er   = el   + (size_t)NNODES * NHEAD;        // N*16
    float* mbuf = er   + (size_t)NNODES * NHEAD;        // N*16
    float* sbuf = mbuf + (size_t)NNODES * NHEAD;        // N*16
    float* stats= sbuf + (size_t)NNODES * NHEAD;        // 256
    float* scl  = stats + 256;                          // 128
    float* shf  = scl + 128;                            // 128

    const int MT = NNODES / 16;                         // 3125 row tiles
    const float invN = 1.0f / (float)NNODES;

    auto bnstats = [&](const float* X, int Cdim, const float* ga, const float* be) {
        zero_stats_kernel<<<1, 256, 0, stream>>>(stats);
        colstats_kernel<<<(NNODES + 127) / 128, Cdim, 0, stream>>>(X, Cdim, stats, NNODES, 128);
        bnfin_kernel<<<1, Cdim, 0, stream>>>(stats, ga, be, invN, scl, shf);
    };

    // ================= embed MLP (64 -> 128, BN+ReLU, 128 -> 128, skip) =====
    gemm_wmma<64, 128, 0, 0><<<MT, 256, 0, stream>>>(x, emb_w1, emb_b1, nullptr, nullptr, tmp);
    bnstats(tmp, HID, emb_ga, emb_be);
    gemm_wmma<128, 128, 1, 1><<<MT, 256, 0, stream>>>(tmp, emb_w2, emb_b2, scl, shf, h);

    // ================= 3 GNN layers =========================================
    for (int l = 0; l < 3; ++l) {
        const float* gb0 = (const float*)d_in[19 + (l * 3 + 0) * 4 + 3];
        const float* gb1 = (const float*)d_in[19 + (l * 3 + 1) * 4 + 3];
        const float* gb2 = (const float*)d_in[19 + (l * 3 + 2) * 4 + 3];
        init_acc_kernel<<<(NNODES * HID) / 256, 256, 0, stream>>>(acc, gb0, gb1, gb2);

        for (int r = 0; r < 3; ++r) {
            const int base = 19 + (l * 3 + r) * 4;
            const float* gW  = (const float*)d_in[base + 0];
            const float* gal = (const float*)d_in[base + 1];
            const float* gar = (const float*)d_in[base + 2];

            gemm_wmma<128, 128, 0, 0><<<MT, 256, 0, stream>>>(h, gW, nullptr, nullptr, nullptr, z);
            elr_kernel<<<(NNODES * NHEAD) / 256, 256, 0, stream>>>(z, gal, gar, el, er);
            init_ms_kernel<<<(NNODES * NHEAD) / 256, 256, 0, stream>>>(mbuf, sbuf);
            scores_max_kernel<<<(NEDGES * NHEAD) / 256, 256, 0, stream>>>(
                src[r], dst[r], el, er, mbuf);
            exp_sum_kernel<<<(NEDGES * NHEAD) / 256, 256, 0, stream>>>(
                src[r], dst[r], el, er, mbuf, sbuf);
            aggregate_kernel<<<NEDGES / 8, 256, 0, stream>>>(
                src[r], dst[r], el, er, mbuf, sbuf, z, acc);
        }

        // per-layer MLP: leaky(acc,0.01) @ w1[128x64] -> BN+ReLU -> @ w2[64x128]
        const int mb = 55 + l * 6;
        const float* m_b1 = (const float*)d_in[mb + 0];
        const float* m_b2 = (const float*)d_in[mb + 1];
        const float* m_be = (const float*)d_in[mb + 2];
        const float* m_ga = (const float*)d_in[mb + 3];
        const float* m_w1 = (const float*)d_in[mb + 4];
        const float* m_w2 = (const float*)d_in[mb + 5];

        gemm_wmma<128, 64, 2, 0><<<MT, 128, 0, stream>>>(acc, m_w1, m_b1, nullptr, nullptr, tmp);
        bnstats(tmp, 64, m_ga, m_be);
        gemm_wmma<64, 128, 1, 2><<<MT, 256, 0, stream>>>(tmp, m_w2, m_b2, scl, shf, h);
    }

    // ================= decision MLP (128 -> 128 BN+ReLU -> 2) ===============
    gemm_wmma<128, 128, 0, 0><<<MT, 256, 0, stream>>>(h, dec_w1, dec_b1, nullptr, nullptr, tmp);
    bnstats(tmp, HID, dec_ga, dec_be);
    decision_kernel<<<(NNODES + 255) / 256, 256, 0, stream>>>(
        tmp, scl, shf, dec_w2, dec_b2, (float*)d_out, NNODES);
}
